// VarlenPatchifier_25254407700573
// MI455X (gfx1250) — matrix operand order repaired and checked by reading, not compiled, not run on hardware
//
#include <hip/hip_runtime.h>

// ---------------------------------------------------------------------------
// VarlenPatchifier for MI455X (gfx1250, wave32, WMMA).
//
// tokens = patchify(images) @ W_proj + b_proj   (M=32768, K=768, N=768)
// fp32 inputs are split into bf16 hi/lo pairs; the GEMM runs on
// v_wmma_f32_16x16x32_bf16 with 3 cross-product passes (hi*hi + hi*lo + lo*hi)
// for near-fp32 accuracy while staying on the matrix pipes and memory-bound
// (~204 MB of traffic -> ~9 us floor at 23.3 TB/s HBM).
// ---------------------------------------------------------------------------

typedef __attribute__((ext_vector_type(16))) __bf16 v16bf;
typedef __attribute__((ext_vector_type(8)))  float  v8f;

#define TOKENS 32768   // B*Hp*Wp = 32*32*32
#define KDIM   768     // C*P*P
#define NDIM   768     // EMBED
#define KT     24      // K tiles of 32
#define MTILES 2048    // TOKENS/16
#define NTILES 48      // NDIM/16

// d_out float offsets (outputs concatenated in reference return order)
#define OFF_CU    25165824ull   // tokens: 32768*768
#define OFF_COORD 25165857ull   // +33
#define OFF_COS   25231393ull   // +65536
#define OFF_SIN   27328545ull   // +2097152
#define OFF_ISP   29425697ull   // +2097152

// ---------------------------------------------------------------------------
// Kernel 1: patchify + fp32 -> bf16 hi/lo split, written in WMMA A-fragment
// layout.  Flat id = ((mtile*24 + kt)*32 + lane)*16 + e
//   row m = mtile*16 + (lane&15)
//   k     = kt*32 + (e<8?0:16) + (lane>>4)*8 + (e&7)     (ISA 16-bit A 16x32)
// ---------------------------------------------------------------------------
__global__ void __launch_bounds__(256) patchify_convert(
    const float* __restrict__ img,
    unsigned short* __restrict__ xh_, unsigned short* __restrict__ xl_)
{
    __bf16* xh = (__bf16*)xh_;
    __bf16* xl = (__bf16*)xl_;
    int id   = blockIdx.x * 256 + threadIdx.x;          // 25,165,824 threads
    int e    = id & 15;
    int lane = (id >> 4) & 31;
    int frag = id >> 9;
    int kt   = frag % KT;
    int mt   = frag / KT;

    int m = (mt << 4) | (lane & 15);
    int k = (kt << 5) + ((e & 8) << 1) + ((lane >> 4) << 3) + (e & 7);

    // m -> (b, hp, wp); k -> (c, p1, p2)
    int b  = m >> 10;
    int rm = m & 1023;
    int hp = rm >> 5, wp = rm & 31;
    int c  = k >> 8;
    int rk = k & 255;
    int p1 = rk >> 4, p2 = rk & 15;

    size_t src = ((size_t)(b * 3 + c) * 512 + (size_t)(hp * 16 + p1)) * 512
               + (size_t)(wp * 16 + p2);
    float v  = img[src];
    __bf16 h = (__bf16)v;
    float  r = v - (float)h;
    xh[id] = h;
    xl[id] = (__bf16)r;
}

// ---------------------------------------------------------------------------
// Kernel 2: W_proj (K x N, row-major) -> bf16 hi/lo in WMMA B-fragment layout.
// Flat id = ((ntile*24 + kt)*32 + lane)*16 + e
//   col n = ntile*16 + (lane&15)
//   k     = kt*32 + (e<8?0:16) + (lane>>4)*8 + (e&7)     (matches A layout)
// ---------------------------------------------------------------------------
__global__ void __launch_bounds__(256) wconvert(
    const float* __restrict__ W,
    unsigned short* __restrict__ wh_, unsigned short* __restrict__ wl_)
{
    __bf16* wh = (__bf16*)wh_;
    __bf16* wl = (__bf16*)wl_;
    int id   = blockIdx.x * 256 + threadIdx.x;          // 589,824 threads
    int e    = id & 15;
    int lane = (id >> 4) & 31;
    int frag = id >> 9;
    int kt   = frag % KT;
    int nt   = frag / KT;

    int n = (nt << 4) | (lane & 15);
    int k = (kt << 5) + ((e & 8) << 1) + ((lane >> 4) << 3) + (e & 7);

    float v  = W[(size_t)k * NDIM + n];
    __bf16 h = (__bf16)v;
    float  r = v - (float)h;
    wh[id] = h;
    wl[id] = (__bf16)r;
}

// ---------------------------------------------------------------------------
// Kernel 3: WMMA GEMM.  Block = 256 threads = 8 waves; wave w owns a 16x64
// output tile (4 accumulators).  grid = (12 N-groups of 64, 256 M-groups of
// 128).  K loop: 24 steps; per step 12 x v_wmma_f32_16x16x32_bf16.
// All fragment loads are one coalesced 32B vector per lane.
// ---------------------------------------------------------------------------
__global__ void __launch_bounds__(256) gemm_wmma(
    const unsigned short* __restrict__ xh_, const unsigned short* __restrict__ xl_,
    const unsigned short* __restrict__ wh_, const unsigned short* __restrict__ wl_,
    const float* __restrict__ bias, float* __restrict__ out)
{
    const __bf16* xh = (const __bf16*)xh_;
    const __bf16* xl = (const __bf16*)xl_;
    const __bf16* wh = (const __bf16*)wh_;
    const __bf16* wl = (const __bf16*)wl_;

    const int lane = threadIdx.x & 31;
    const int wave = threadIdx.x >> 5;
    const int mt   = blockIdx.y * 8 + wave;   // 16-row tile index, 0..2047
    const int ng   = blockIdx.x;              // 64-col group, 0..11

    v8f acc[4] = {};

    const size_t aoff = ((size_t)mt * KT * 32 + lane) * 16;
    const __bf16* ahp = xh + aoff;
    const __bf16* alp = xl + aoff;
    const size_t boff = ((size_t)(ng * 4) * KT * 32 + lane) * 16;

    for (int kt = 0; kt < KT; ++kt) {
        v16bf ah = *(const v16bf*)(ahp + (size_t)kt * 512);
        v16bf al = *(const v16bf*)(alp + (size_t)kt * 512);
#pragma unroll
        for (int j = 0; j < 4; ++j) {
            const size_t bo = boff + ((size_t)j * KT + kt) * 512;
            v16bf bh = *(const v16bf*)(wh + bo);
            v16bf bl = *(const v16bf*)(wl + bo);
            acc[j] = __builtin_amdgcn_wmma_f32_16x16x32_bf16(
                false, ah, false, bh, (short)0, acc[j], false, false);
            acc[j] = __builtin_amdgcn_wmma_f32_16x16x32_bf16(
                false, ah, false, bl, (short)0, acc[j], false, false);
            acc[j] = __builtin_amdgcn_wmma_f32_16x16x32_bf16(
                false, al, false, bh, (short)0, acc[j], false, false);
        }
    }

    // C/D layout: VGPR r -> row r + 8*(lane/16); col = lane%16
    const int half = lane >> 4;
    const int col0 = lane & 15;
    const int row0 = mt * 16 + 8 * half;
#pragma unroll
    for (int j = 0; j < 4; ++j) {
        const int col = ng * 64 + j * 16 + col0;
        const float bv = bias[col];
#pragma unroll
        for (int r = 0; r < 8; ++r) {
            out[(size_t)(row0 + r) * NDIM + col] = acc[j][r] + bv;
        }
    }
}

// ---------------------------------------------------------------------------
// Kernel 4: auxiliary outputs (cu_seqlens, patch_coords, rope cos/sin,
// is_patch), all written as f32 into the concatenated output buffer.
// ---------------------------------------------------------------------------
__global__ void __launch_bounds__(256) aux_outputs(float* __restrict__ out)
{
    int t = blockIdx.x * 256 + threadIdx.x;             // 32768 tokens
    int idx = t & 1023;
    int ys = idx >> 5, xs = idx & 31;

    out[OFF_COORD + 2 * (size_t)t]     = (float)ys;
    out[OFF_COORD + 2 * (size_t)t + 1] = (float)xs;
    out[OFF_ISP + t] = 1.0f;
    if (t < 33) out[OFF_CU + t] = (float)(t * 1024);

    const float LOG_BASE = 9.2103403719761836f;         // ln(10000)
    size_t base = (size_t)t * 64;
#pragma unroll
    for (int i = 0; i < 16; ++i) {
        float invf = __expf(-((float)i / 16.0f) * LOG_BASE);  // 10000^(-2i/32)
        float ay = (float)ys * invf;
        float ax = (float)xs * invf;
        float cy = __cosf(ay), sy = __sinf(ay);
        float cx = __cosf(ax), sx = __sinf(ax);
        out[OFF_COS + base + 2 * i]          = cy;
        out[OFF_COS + base + 2 * i + 1]      = cy;
        out[OFF_COS + base + 32 + 2 * i]     = cx;
        out[OFF_COS + base + 32 + 2 * i + 1] = cx;
        out[OFF_SIN + base + 2 * i]          = sy;
        out[OFF_SIN + base + 2 * i + 1]      = sy;
        out[OFF_SIN + base + 32 + 2 * i]     = sx;
        out[OFF_SIN + base + 32 + 2 * i + 1] = sx;
    }
}

// ---------------------------------------------------------------------------
extern "C" void kernel_launch(void* const* d_in, const int* in_sizes, int n_in,
                              void* d_out, int out_size, void* d_ws, size_t ws_size,
                              hipStream_t stream)
{
    const float* images = (const float*)d_in[0];
    const float* W_proj = (const float*)d_in[1];
    const float* b_proj = (const float*)d_in[2];
    float* out = (float*)d_out;

    const size_t XN = (size_t)TOKENS * KDIM;   // 25,165,824 elems
    const size_t WN = (size_t)KDIM * NDIM;     //    589,824 elems
    const size_t need = (2 * XN + 2 * WN) * sizeof(unsigned short); // ~98.3 MB
    if (ws_size < need) return;                // deterministic guard

    unsigned short* xh = (unsigned short*)d_ws;
    unsigned short* xl = xh + XN;
    unsigned short* wh = xl + XN;
    unsigned short* wl = wh + WN;

    patchify_convert<<<(int)(XN / 256), 256, 0, stream>>>(images, xh, xl);
    wconvert<<<(int)(WN / 256), 256, 0, stream>>>(W_proj, wh, wl);

    dim3 grid(NDIM / 64, MTILES / 8);          // (12, 256)
    gemm_wmma<<<grid, 256, 0, stream>>>(xh, xl, wh, wl, b_proj, out);

    aux_outputs<<<TOKENS / 256, 256, 0, stream>>>(out);
}
